// GATLayer_996432413149
// MI455X (gfx1250) — matrix operand order repaired and checked by reading, not compiled, not run on hardware
//
#include <hip/hip_runtime.h>
#include <cstdint>

typedef __attribute__((ext_vector_type(16))) __bf16 v16bf;
typedef __attribute__((ext_vector_type(8)))  float  v8f;

#define B_    8
#define C_    2048
#define FIN_  128
#define FOUT_ 64
#define NEG_SLOPE 0.2f

struct U8 { uint32_t u[8]; };
struct BU { uint4 lo; uint4 hi; };     // one 32x16 bf16 B fragment (8 VGPRs)

__device__ __forceinline__ uint32_t f2bf1(float f) {
  uint32_t u = __float_as_uint(f);
  return (u + 0x7FFFu + ((u >> 16) & 1u)) >> 16;   // round-to-nearest-even bf16
}
__device__ __forceinline__ uint32_t packbf(float a, float b) {
  return f2bf1(a) | (f2bf1(b) << 16);
}
__device__ __forceinline__ float lrelu(float x) { return x > 0.f ? x : NEG_SLOPE * x; }

// ---------------------------------------------------------------------------
// Kernel 1: Wh = H @ W (f32), s1 = Wh@a1, s2 = Wh@a2.
// Block: 128 threads = 2 rows x 64 output features.
// ---------------------------------------------------------------------------
__global__ void __launch_bounds__(128) wh_s_kernel(
    const float* __restrict__ H, const float* __restrict__ W,
    const float* __restrict__ a, float* __restrict__ Wh,
    float* __restrict__ s1, float* __restrict__ s2) {
  __shared__ float hs[2 * FIN_];
  __shared__ float r1[128], r2[128];
  int t = threadIdx.x;
  int row0 = blockIdx.x * 2;                       // flat row = b*C + c
  hs[t]        = H[(size_t)row0 * FIN_ + t];
  hs[t + 128]  = H[(size_t)row0 * FIN_ + 128 + t];
  __syncthreads();
  int lr = t >> 6;                                 // local row 0..1
  int f  = t & 63;
  const float* hrow = &hs[lr * FIN_];
  float acc = 0.f;
#pragma unroll 8
  for (int k = 0; k < FIN_; ++k) acc += hrow[k] * W[k * FOUT_ + f];
  int grow = row0 + lr;
  Wh[(size_t)grow * FOUT_ + f] = acc;
  r1[t] = acc * a[f];
  r2[t] = acc * a[FOUT_ + f];
  __syncthreads();
  for (int s = 32; s >= 1; s >>= 1) {              // reduce each 64-thread half
    if (f < s) { r1[t] += r1[t + s]; r2[t] += r2[t + s]; }
    __syncthreads();
  }
  if (f == 0) { s1[grow] = r1[t]; s2[grow] = r2[t]; }
}

// ---------------------------------------------------------------------------
// Kernel 2: pre-swizzle Wh into bf16 WMMA B-fragments.
// Layout: Whb[b][chunk(64)][ntile(4)][lane(32)][v(8)] dwords.
// B matrix 32x16 bf16 layout: lane L: n = L&15, h = L>>4; dword v packs
// rows K = (v>>2)*16 + h*8 + 2(v&3) and that + 1 of column n.
// ---------------------------------------------------------------------------
__global__ void __launch_bounds__(128) pack_kernel(
    const float* __restrict__ Wh, uint32_t* __restrict__ Whb) {
  int g = blockIdx.x * 128 + threadIdx.x;          // 0 .. 524287
  int v     =  g        & 7;
  int lane  = (g >> 3)  & 31;
  int nt    = (g >> 8)  & 3;
  int chunk = (g >> 10) & 63;
  int b     =  g >> 16;
  int h = lane >> 4;
  int n = nt * 16 + (lane & 15);
  int k0 = ((v >> 2) << 4) + (h << 3) + ((v & 3) << 1);
  int row = chunk * 32 + k0;                       // c index (K dim)
  size_t base = ((size_t)b * C_ + row) * FOUT_ + n;
  Whb[g] = packbf(Wh[base], Wh[base + FOUT_]);
}

// ---------------------------------------------------------------------------
// Kernel 3: fused masked-softmax attention + P@Wh via bf16 WMMA.
// 1 block = (batch b, 16-row i-tile). 4 waves split the 2048 j-columns.
// Two-pass exact softmax: LReLU is monotone, so the row max of e is
// LReLU(s1[i] + max_{allowed j} s2[j]) -> no online rescaling needed, and
// partial accumulators across waves combine by plain addition.
// ---------------------------------------------------------------------------
__global__ void __launch_bounds__(128) gat_kernel(
    const int* __restrict__ A, const float* __restrict__ s1g,
    const float* __restrict__ s2g, const uint32_t* __restrict__ Whb,
    float* __restrict__ out) {
  __shared__ float s2s[C_];          // 8 KB: s2 row for this batch
  __shared__ float red[128];
  __shared__ float mrow[16];
  __shared__ float ltot[16];
  __shared__ float accs[4096];       // 16 KB: per-wave partial acc tiles

  int t = threadIdx.x;
  int w = t >> 5, lane = t & 31;
  int blk = blockIdx.x;
  int b  = blk >> 7;
  int i0 = (blk & 127) << 4;
  const int*   Ab  = A   + (size_t)b * C_ * C_;
  const float* s2b = s2g + (size_t)b * C_;

  for (int i = t; i < C_; i += 128) s2s[i] = s2b[i];
  __syncthreads();

  // ---- Pass A: per-row max of s2 over allowed columns (A>0 | diagonal) ----
  {
    int r = t >> 3, q = t & 7;                     // 16 rows x 8 threads
    float vmax = -3.0e38f;
    if (q == 0) vmax = s2s[i0 + r];                // diagonal always allowed
    const int4* ap = (const int4*)(Ab + (size_t)(i0 + r) * C_ + q * 256);
#pragma unroll 4
    for (int jj = 0; jj < 64; ++jj) {
      int4 av = ap[jj];
      int jb = q * 256 + jj * 4;
      if (av.x > 0) vmax = fmaxf(vmax, s2s[jb + 0]);
      if (av.y > 0) vmax = fmaxf(vmax, s2s[jb + 1]);
      if (av.z > 0) vmax = fmaxf(vmax, s2s[jb + 2]);
      if (av.w > 0) vmax = fmaxf(vmax, s2s[jb + 3]);
    }
    red[t] = vmax;
  }
  __syncthreads();
  if (t < 16) {
    float m = red[t * 8];
#pragma unroll
    for (int q = 1; q < 8; ++q) m = fmaxf(m, red[t * 8 + q]);
    mrow[t] = lrelu(s1g[(size_t)b * C_ + i0 + t] + m);
  }
  __syncthreads();

  // ---- Pass B: build P tiles (bf16), WMMA-accumulate P @ Wh (L2 hits on A) --
  int h = lane >> 4, m = lane & 15;
  int gi = i0 + m;
  const int* Arow = Ab + (size_t)gi * C_;
  float s1m = s1g[(size_t)b * C_ + gi];
  float mr  = mrow[m];
  v8f accv[4];
  accv[0] = {}; accv[1] = {}; accv[2] = {}; accv[3] = {};
  float lsum = 0.f;
  const uint32_t* WhbB = Whb + (size_t)b * 64 * 4 * 32 * 8;

  for (int c = w * 16; c < w * 16 + 16; ++c) {     // 16 K-chunks of 32 per wave
    int j0 = c * 32;

    // ---- issue ALL VMEM for this chunk up front: A mask first (waited for
    //      by the P build at loadcnt<=8), then the 4 B fragments (waited for
    //      only at the WMMAs, hidden under the exp/pack VALU work).
    int2 av[8];
#pragma unroll
    for (int v = 0; v < 8; ++v) {
      int k0 = ((v >> 2) << 4) + (h << 3) + ((v & 3) << 1);
      av[v] = *(const int2*)(Arow + j0 + k0);      // mask bits (L2 resident)
    }
    const uint32_t* fb = WhbB + (size_t)c * 1024 + lane * 8;
    BU bu[4];
#pragma unroll
    for (int nt = 0; nt < 4; ++nt) {
      bu[nt].lo = *(const uint4*)(fb + nt * 256);
      bu[nt].hi = *(const uint4*)(fb + nt * 256 + 4);
    }
    __builtin_amdgcn_sched_barrier(0);   // keep the load clause above

    // ---- VALU: masked exp, bf16 pack (A-matrix 16x32 bf16 WMMA layout) ----
    U8 au;
    float lacc = 0.f;
#pragma unroll
    for (int v = 0; v < 8; ++v) {
      int k0 = ((v >> 2) << 4) + (h << 3) + ((v & 3) << 1);
      int j = j0 + k0;
      float2 sv = *(const float2*)(&s2s[j]);
      float e0 = lrelu(s1m + sv.x);
      float e1 = lrelu(s1m + sv.y);
      bool ok0 = (av[v].x > 0) || (gi == j);
      bool ok1 = (av[v].y > 0) || (gi == j + 1);
      float p0 = ok0 ? __expf(e0 - mr) : 0.f;
      float p1 = ok1 ? __expf(e1 - mr) : 0.f;
      lacc += p0 + p1;
      au.u[v] = packbf(p0, p1);
    }
    lsum += lacc;
    v16bf afrag = __builtin_bit_cast(v16bf, au);
    __builtin_amdgcn_sched_barrier(0);   // keep P build under the B loads

    // ---- matrix pipe: 4 back-to-back WMMAs (N = 64 split into 4 tiles) ----
#pragma unroll
    for (int nt = 0; nt < 4; ++nt) {
      v16bf bfrag = __builtin_bit_cast(v16bf, bu[nt]);
      accv[nt] = __builtin_amdgcn_wmma_f32_16x16x32_bf16(
          false, afrag, false, bfrag, (short)0, accv[nt], false, false);
    }
  }

  // ---- combine 4 waves' partials, normalize, relu, store ----
#pragma unroll
  for (int nt = 0; nt < 4; ++nt)
#pragma unroll
    for (int r = 0; r < 8; ++r)
      accs[w * 1024 + nt * 256 + r * 32 + lane] = accv[nt][r];
  red[t] = lsum;
  __syncthreads();
  if (t < 16) {
    float L = 0.f;
#pragma unroll
    for (int ww = 0; ww < 4; ++ww) L += red[ww * 32 + t] + red[ww * 32 + 16 + t];
    ltot[t] = L;
  }
  __syncthreads();
  {
    int col = t & 63;
    int nt = col >> 4, nn = col & 15;
#pragma unroll
    for (int mm = (t >> 6); mm < 16; mm += 2) {
      int r = mm & 7, hh = mm >> 3;
      int ln = hh * 16 + nn;
      float sum = 0.f;
#pragma unroll
      for (int ww = 0; ww < 4; ++ww) sum += accs[ww * 1024 + nt * 256 + r * 32 + ln];
      float val = sum / ltot[mm];
      out[(((size_t)b * C_) + (i0 + mm)) * FOUT_ + col] = fmaxf(val, 0.f);
    }
  }
}

// ---------------------------------------------------------------------------
extern "C" void kernel_launch(void* const* d_in, const int* in_sizes, int n_in,
                              void* d_out, int out_size, void* d_ws, size_t ws_size,
                              hipStream_t stream) {
  const float* H = (const float*)d_in[0];
  const int*   A = (const int*)d_in[1];
  const float* W = (const float*)d_in[2];
  const float* a = (const float*)d_in[3];
  float* out = (float*)d_out;

  char* ws = (char*)d_ws;
  float*    Wh  = (float*)ws;                                   // 4 MiB
  float*    s1  = (float*)(ws + (4 << 20));                     // 64 KiB
  float*    s2  = (float*)(ws + (4 << 20) + (64 << 10));        // 64 KiB
  uint32_t* Whb = (uint32_t*)(ws + (4 << 20) + (128 << 10));    // 2 MiB

  wh_s_kernel<<<dim3(B_ * C_ / 2), dim3(128), 0, stream>>>(H, W, a, Wh, s1, s2);
  pack_kernel<<<dim3(B_ * 64 * 4 * 32 * 8 / 128), dim3(128), 0, stream>>>(Wh, Whb);
  gat_kernel<<<dim3(B_ * (C_ / 16)), dim3(128), 0, stream>>>(A, s1, s2, Whb, out);
}
